// AvgSeerMambaTransLayer_50637664420399
// MI455X (gfx1250) — matrix operand order repaired and checked by reading, not compiled, not run on hardware
//
#include <hip/hip_runtime.h>

#define C_DIM   256
#define DMODEL  512
#define DINNER  1024
#define DSTATE  256
#define DTRANK  32
#define DCONV   4
#define BATCH   2
#define SEQ     512
#define MROWS   (BATCH * SEQ)      /* 1024 */
#define LDDBL   (DTRANK + 2 * DSTATE) /* 544 */
#define LN_EPS  1e-5f

typedef __bf16 bf16_t;
typedef __attribute__((ext_vector_type(8)))  __bf16 v8bf;
typedef __attribute__((ext_vector_type(16))) __bf16 v16bf;
typedef __attribute__((ext_vector_type(8)))  float  v8f;

// ---------------------------------------------------------------- utilities
__global__ void f32_to_bf16_kernel(const float* __restrict__ in,
                                   bf16_t* __restrict__ out, int n) {
  int i = blockIdx.x * blockDim.x + threadIdx.x;
  if (i < n) out[i] = (bf16_t)in[i];
}

// mean over sequence for each (b, c)
__global__ void avg_kernel(const float* __restrict__ x, float* __restrict__ avg) {
  int gid = blockIdx.x * blockDim.x + threadIdx.x;
  if (gid >= BATCH * C_DIM) return;
  int b = gid / C_DIM, c = gid % C_DIM;
  float s = 0.f;
  for (int l = 0; l < SEQ; ++l) s += x[(b * SEQ + l) * C_DIM + c];
  avg[gid] = s * (1.0f / SEQ);
}

// concat [x, avg] -> layernorm(512) -> bf16 rows for in_proj GEMM
__global__ void ln1_pack_kernel(const float* __restrict__ x,
                                const float* __restrict__ avg,
                                const float* __restrict__ w,
                                const float* __restrict__ bch,
                                bf16_t* __restrict__ u) {
  __shared__ float red[256];
  int row = blockIdx.x;              // 0..MROWS-1
  int b = row / SEQ;
  int tid = threadIdx.x;             // 0..255, covers 2 columns each
  float v0 = x[row * C_DIM + tid];         // col = tid        (< 256)
  float v1 = avg[b * C_DIM + tid];         // col = 256 + tid
  red[tid] = v0 + v1;
  __syncthreads();
  for (int s = 128; s > 0; s >>= 1) { if (tid < s) red[tid] += red[tid + s]; __syncthreads(); }
  float mu = red[0] * (1.0f / DMODEL);
  __syncthreads();
  float d0 = v0 - mu, d1 = v1 - mu;
  red[tid] = d0 * d0 + d1 * d1;
  __syncthreads();
  for (int s = 128; s > 0; s >>= 1) { if (tid < s) red[tid] += red[tid + s]; __syncthreads(); }
  float rs = rsqrtf(red[0] * (1.0f / DMODEL) + LN_EPS);
  u[row * DMODEL + tid]       = (bf16_t)(d0 * rs * w[tid]       + bch[tid]);
  u[row * DMODEL + 256 + tid] = (bf16_t)(d1 * rs * w[256 + tid] + bch[256 + tid]);
}

// ---------------------------------------------------------------- WMMA GEMM
// C[M,N] = A[M,K](bf16, row-major) * Bt[N,K](bf16, row-major)^T  (f32 accum)
// epilogue: (+bias[n]) -> (softplus if act==1) -> (+resid[m*N+n]) -> store f32
__device__ __forceinline__ v16bf load_frag_a(const bf16_t* __restrict__ p) {
  // A 16x32 bf16 fragment: 16 contiguous K values per half-wave
  const v8bf* q = (const v8bf*)p;
  v8bf lo = q[0], hi = q[1];
  return __builtin_shufflevector(lo, hi, 0,1,2,3,4,5,6,7,8,9,10,11,12,13,14,15);
}
__device__ __forceinline__ v16bf load_frag_b(const bf16_t* __restrict__ p) {
  // B 32x16 bf16 fragment: 16 contiguous K values per half-wave
  const v8bf* q = (const v8bf*)p;
  v8bf lo = q[0], hi = q[1];
  return __builtin_shufflevector(lo, hi, 0,1,2,3,4,5,6,7,8,9,10,11,12,13,14,15);
}

__global__ __launch_bounds__(256)
void gemm_bf16_wmma(const bf16_t* __restrict__ A, const bf16_t* __restrict__ Bt,
                    float* __restrict__ Cout, int M, int N, int K,
                    const float* __restrict__ bias,
                    const float* __restrict__ resid, int act) {
  const int lane = threadIdx.x & 31;
  const int wave = threadIdx.x >> 5;        // 8 waves / block
  const int m0 = blockIdx.y * 64 + (wave >> 2) * 32;   // wave: 2 M-tiles
  const int n0 = blockIdx.x * 128 + (wave & 3) * 32;   // wave: 2 N-tiles
  const int lr = lane & 15;
  const bool nv0 = (n0 < N), nv1 = (n0 + 16 < N);

  const bf16_t* arow0 = A + (size_t)(m0 + lr) * K + ((lane & 16) ? 16 : 0);
  const bf16_t* arow1 = arow0 + (size_t)16 * K;
  const bf16_t* brow0 = Bt + (size_t)((nv0 ? n0      : 0) + lr) * K + ((lane & 16) ? 16 : 0);
  const bf16_t* brow1 = Bt + (size_t)((nv1 ? n0 + 16 : 0) + lr) * K + ((lane & 16) ? 16 : 0);

  v8f acc00 = {}, acc01 = {}, acc10 = {}, acc11 = {};
  for (int k0 = 0; k0 < K; k0 += 32) {
    __builtin_prefetch(arow0 + k0 + 64, 0, 1);   // global_prefetch_b8 two k-tiles ahead
    __builtin_prefetch(brow0 + k0 + 64, 0, 1);
    v16bf a0 = load_frag_a(arow0 + k0);
    v16bf a1 = load_frag_a(arow1 + k0);
    v16bf b0 = load_frag_b(brow0 + k0);
    v16bf b1 = load_frag_b(brow1 + k0);
    acc00 = __builtin_amdgcn_wmma_f32_16x16x32_bf16(false, a0, false, b0, (short)0, acc00, false, false);
    acc01 = __builtin_amdgcn_wmma_f32_16x16x32_bf16(false, a0, false, b1, (short)0, acc01, false, false);
    acc10 = __builtin_amdgcn_wmma_f32_16x16x32_bf16(false, a1, false, b0, (short)0, acc10, false, false);
    acc11 = __builtin_amdgcn_wmma_f32_16x16x32_bf16(false, a1, false, b1, (short)0, acc11, false, false);
  }

  const int crow = (lane >> 4) << 3;  // 0 or 8 (C/D layout)
  auto store_tile = [&](v8f acc, int mt, int nt, bool valid) {
    if (!valid) return;
#pragma unroll
    for (int r = 0; r < 8; ++r) {
      int m = mt + crow + r;
      int n = nt + lr;
      float v = acc[r];
      if (bias)  v += bias[n];
      if (act == 1) v = (v > 20.f) ? v : log1pf(__expf(v));   // softplus
      if (resid) v += resid[(size_t)m * N + n];
      Cout[(size_t)m * N + n] = v;
    }
  };
  store_tile(acc00, m0,      n0,      nv0);
  store_tile(acc01, m0,      n0 + 16, nv1);
  store_tile(acc10, m0 + 16, n0,      nv0);
  store_tile(acc11, m0 + 16, n0 + 16, nv1);
}

// ---------------------------------------------------------------- conv+SiLU
__global__ void conv_silu_kernel(const float* __restrict__ xz,
                                 const float* __restrict__ cw,
                                 const float* __restrict__ cb,
                                 float* __restrict__ xc,
                                 bf16_t* __restrict__ xcb) {
  int idx = blockIdx.x * blockDim.x + threadIdx.x;
  if (idx >= MROWS * DINNER) return;
  int m = idx / DINNER, d = idx % DINNER;
  int b = m / SEQ, l = m % SEQ;
  float acc = cb[d];
#pragma unroll
  for (int k = 0; k < DCONV; ++k) {
    int lk = l + k - (DCONV - 1);
    if (lk >= 0) acc += xz[(size_t)(b * SEQ + lk) * (2 * DINNER) + d] * cw[d * DCONV + k];
  }
  float s = acc / (1.f + __expf(-acc));   // SiLU
  xc[idx] = s;
  xcb[idx] = (bf16_t)s;
}

// extract dt columns (0..31) of dbl as bf16 for dt_proj GEMM
__global__ void dt_extract_kernel(const float* __restrict__ dbl,
                                  bf16_t* __restrict__ dtin) {
  int idx = blockIdx.x * blockDim.x + threadIdx.x;
  if (idx >= MROWS * DTRANK) return;
  int m = idx / DTRANK, r = idx % DTRANK;
  dtin[idx] = (bf16_t)dbl[(size_t)m * LDDBL + r];
}

// ---------------------------------------------------------------- selective scan
// one wave32 per channel d; each lane owns 8 of the 256 states in registers
__global__ __launch_bounds__(256)
void scan_kernel(const float* __restrict__ dbl, const float* __restrict__ dtf,
                 const float* __restrict__ xc, const float* __restrict__ A_log,
                 float* __restrict__ ys) {
  __shared__ float Bsh[DSTATE];
  __shared__ float Csh[DSTATE];
  const int lane = threadIdx.x & 31;
  const int wave = threadIdx.x >> 5;
  const int b = blockIdx.x >> 7;                 // 128 blocks per batch
  const int d = (blockIdx.x & 127) * 8 + wave;   // 8 channels per block
  float Ad[8], h[8];
#pragma unroll
  for (int j = 0; j < 8; ++j) {
    Ad[j] = -__expf(A_log[(size_t)d * DSTATE + lane + 32 * j]);  // A = -exp(A_log)
    h[j] = 0.f;
  }
  for (int t = 0; t < SEQ; ++t) {
    const int row = b * SEQ + t;
    __syncthreads();
    Bsh[threadIdx.x] = dbl[(size_t)row * LDDBL + DTRANK + threadIdx.x];
    Csh[threadIdx.x] = dbl[(size_t)row * LDDBL + DTRANK + DSTATE + threadIdx.x];
    __syncthreads();
    float dt  = dtf[(size_t)row * DINNER + d];
    float dtx = dt * xc[(size_t)row * DINNER + d];
    float y = 0.f;
#pragma unroll
    for (int j = 0; j < 8; ++j) {
      float dA = __expf(dt * Ad[j]);
      h[j] = dA * h[j] + dtx * Bsh[lane + 32 * j];
      y += h[j] * Csh[lane + 32 * j];
    }
#pragma unroll
    for (int off = 16; off > 0; off >>= 1) y += __shfl_xor(y, off, 32);
    if (lane == 0) ys[(size_t)row * DINNER + d] = y;
  }
}

// y = (ys + xc*D_skip) * silu(z)  -> bf16 for out_proj GEMM
__global__ void gate_kernel(const float* __restrict__ ys, const float* __restrict__ xc,
                            const float* __restrict__ xz, const float* __restrict__ Dskip,
                            bf16_t* __restrict__ yb) {
  int idx = blockIdx.x * blockDim.x + threadIdx.x;
  if (idx >= MROWS * DINNER) return;
  int m = idx / DINNER, d = idx % DINNER;
  float z = xz[(size_t)m * (2 * DINNER) + DINNER + d];
  float g = z / (1.f + __expf(-z));
  yb[idx] = (bf16_t)((ys[idx] + xc[idx] * Dskip[d]) * g);
}

// layernorm(256) of residual stream -> bf16 for head GEMM
__global__ void ln2_kernel(const float* __restrict__ hin, const float* __restrict__ w,
                           const float* __restrict__ bch, bf16_t* __restrict__ out) {
  __shared__ float red[256];
  int row = blockIdx.x, tid = threadIdx.x;
  float v = hin[(size_t)row * C_DIM + tid];
  red[tid] = v;
  __syncthreads();
  for (int s = 128; s > 0; s >>= 1) { if (tid < s) red[tid] += red[tid + s]; __syncthreads(); }
  float mu = red[0] * (1.0f / C_DIM);
  __syncthreads();
  float dv = v - mu;
  red[tid] = dv * dv;
  __syncthreads();
  for (int s = 128; s > 0; s >>= 1) { if (tid < s) red[tid] += red[tid + s]; __syncthreads(); }
  float rs = rsqrtf(red[0] * (1.0f / C_DIM) + LN_EPS);
  out[(size_t)row * C_DIM + tid] = (bf16_t)(dv * rs * w[tid] + bch[tid]);
}

// ---------------------------------------------------------------- launch
extern "C" void kernel_launch(void* const* d_in, const int* in_sizes, int n_in,
                              void* d_out, int out_size, void* d_ws, size_t ws_size,
                              hipStream_t stream) {
  const float* x          = (const float*)d_in[0];
  const float* ln1_w      = (const float*)d_in[1];
  const float* ln1_b      = (const float*)d_in[2];
  const float* ln2_w      = (const float*)d_in[3];
  const float* ln2_b      = (const float*)d_in[4];
  const float* head_w     = (const float*)d_in[5];
  const float* head_b     = (const float*)d_in[6];
  const float* in_proj_w  = (const float*)d_in[7];
  const float* conv_w     = (const float*)d_in[8];
  const float* conv_b     = (const float*)d_in[9];
  const float* x_proj_w   = (const float*)d_in[10];
  const float* dt_proj_w  = (const float*)d_in[11];
  const float* dt_proj_b  = (const float*)d_in[12];
  const float* A_log      = (const float*)d_in[13];
  const float* D_skip     = (const float*)d_in[14];
  const float* out_proj_w = (const float*)d_in[15];
  float* out = (float*)d_out;

  char* wsp = (char*)d_ws;
  auto carve = [&](size_t bytes) -> void* {
    void* p = (void*)wsp;
    wsp += (bytes + 255) & ~(size_t)255;
    return p;
  };
  float*  avg    = (float*) carve((size_t)BATCH * C_DIM * 4);
  bf16_t* u_bf   = (bf16_t*)carve((size_t)MROWS * DMODEL * 2);
  float*  xz     = (float*) carve((size_t)MROWS * 2 * DINNER * 4);
  float*  xc     = (float*) carve((size_t)MROWS * DINNER * 4);
  bf16_t* xc_bf  = (bf16_t*)carve((size_t)MROWS * DINNER * 2);
  float*  dbl    = (float*) carve((size_t)MROWS * LDDBL * 4);
  bf16_t* dt_in  = (bf16_t*)carve((size_t)MROWS * DTRANK * 2);
  float*  dtf    = (float*) carve((size_t)MROWS * DINNER * 4);
  float*  ysb    = (float*) carve((size_t)MROWS * DINNER * 4);
  bf16_t* y_bf   = (bf16_t*)carve((size_t)MROWS * DINNER * 2);
  float*  hres   = (float*) carve((size_t)MROWS * C_DIM * 4);
  bf16_t* ln2_bf = (bf16_t*)carve((size_t)MROWS * C_DIM * 2);
  bf16_t* w_in   = (bf16_t*)carve((size_t)2 * DINNER * DMODEL * 2);
  bf16_t* w_xp   = (bf16_t*)carve((size_t)LDDBL * DINNER * 2);
  bf16_t* w_dt   = (bf16_t*)carve((size_t)DINNER * DTRANK * 2);
  bf16_t* w_op   = (bf16_t*)carve((size_t)C_DIM * DINNER * 2);
  bf16_t* w_hd   = (bf16_t*)carve((size_t)C_DIM * C_DIM * 2);

  const int T = 256;
  auto cdiv = [](int a, int b) { return (a + b - 1) / b; };

  // weight conversions f32 -> bf16
  { int n = 2 * DINNER * DMODEL; f32_to_bf16_kernel<<<cdiv(n, T), T, 0, stream>>>(in_proj_w,  w_in, n); }
  { int n = LDDBL * DINNER;      f32_to_bf16_kernel<<<cdiv(n, T), T, 0, stream>>>(x_proj_w,   w_xp, n); }
  { int n = DINNER * DTRANK;     f32_to_bf16_kernel<<<cdiv(n, T), T, 0, stream>>>(dt_proj_w,  w_dt, n); }
  { int n = C_DIM * DINNER;      f32_to_bf16_kernel<<<cdiv(n, T), T, 0, stream>>>(out_proj_w, w_op, n); }
  { int n = C_DIM * C_DIM;       f32_to_bf16_kernel<<<cdiv(n, T), T, 0, stream>>>(head_w,     w_hd, n); }

  // 1. sequence mean + 2. concat/layernorm pack
  avg_kernel<<<cdiv(BATCH * C_DIM, T), T, 0, stream>>>(x, avg);
  ln1_pack_kernel<<<MROWS, T, 0, stream>>>(x, avg, ln1_w, ln1_b, u_bf);

  // 3. in_proj: [1024,512] x [2048,512]^T -> xz
  gemm_bf16_wmma<<<dim3(2 * DINNER / 128, MROWS / 64), T, 0, stream>>>(
      u_bf, w_in, xz, MROWS, 2 * DINNER, DMODEL, nullptr, nullptr, 0);

  // 4. depthwise causal conv + SiLU
  conv_silu_kernel<<<cdiv(MROWS * DINNER, T), T, 0, stream>>>(xz, conv_w, conv_b, xc, xc_bf);

  // 5. x_proj: [1024,1024] x [544,1024]^T -> dbl (dt|B|C)
  gemm_bf16_wmma<<<dim3(cdiv(LDDBL, 128), MROWS / 64), T, 0, stream>>>(
      xc_bf, w_xp, dbl, MROWS, LDDBL, DINNER, nullptr, nullptr, 0);

  // 6. dt_proj + softplus: [1024,32] x [1024,32]^T (+bias, softplus) -> dtf
  dt_extract_kernel<<<cdiv(MROWS * DTRANK, T), T, 0, stream>>>(dbl, dt_in);
  gemm_bf16_wmma<<<dim3(DINNER / 128, MROWS / 64), T, 0, stream>>>(
      dt_in, w_dt, dtf, MROWS, DINNER, DTRANK, dt_proj_b, nullptr, 1);

  // 7. selective scan (sequential over SEQ; wave per channel)
  scan_kernel<<<BATCH * (DINNER / 8), T, 0, stream>>>(dbl, dtf, xc, A_log, ysb);

  // 8. skip + gate -> bf16
  gate_kernel<<<cdiv(MROWS * DINNER, T), T, 0, stream>>>(ysb, xc, xz, D_skip, y_bf);

  // 9. out_proj + residual x: [1024,1024] x [256,1024]^T + x -> hres
  gemm_bf16_wmma<<<dim3(C_DIM / 128, MROWS / 64), T, 0, stream>>>(
      y_bf, w_op, hres, MROWS, C_DIM, DINNER, nullptr, x, 0);

  // 10. layernorm + 11. head GEMM + bias + residual -> out
  ln2_kernel<<<MROWS, T, 0, stream>>>(hres, ln2_w, ln2_b, ln2_bf);
  gemm_bf16_wmma<<<dim3(C_DIM / 128, MROWS / 64), T, 0, stream>>>(
      ln2_bf, w_hd, out, MROWS, C_DIM, C_DIM, head_b, hres, 0);
}
